// MockBenchmarkModel_6562710028728
// MI455X (gfx1250) — matrix operand order repaired and checked by reading, not compiled, not run on hardware
//
#include <hip/hip_runtime.h>
#include <hip/hip_bf16.h>
#include <math.h>

#define Tn 2048
#define Hh 1024
#define Ee 8
#define Ii 4096
#define Vv 32000
#define MAX_T32 136   // ceil(4096/32) + 8 experts of padding slack
#define STR 40        // LDS row stride (u16); rows 80B -> 16B-aligned for b128 async

// ws int-region layout (indices into int* ints)
//  [0..8)     expert counts
//  [8..17)    32-padded segment offsets (seg[8] = total rows)
//  [17]       ntiles
//  [32..168)  tile expert id
//  [296..432) tile row base (global padded row index)
//  [560..696) tile valid row count (1..32)
//  [1024..5120) per-(token,k) slot within expert
//  [5120..9472) token list, grouped by padded expert segment

typedef __attribute__((ext_vector_type(16))) __bf16 v16bf;
typedef __attribute__((ext_vector_type(8)))  float  v8f;

__device__ __forceinline__ unsigned short f2bf(float f) {
  unsigned int u = __float_as_uint(f);
  u += 0x7FFFu + ((u >> 16) & 1u);   // round-to-nearest-even
  return (unsigned short)(u >> 16);
}

// Async global->LDS copy of 16B (CDNA5 TDM-class path, tracked by ASYNCcnt).
// LDS address = low 32 bits of generic pointer (LDS aperture maps addr[31:0]).
__device__ __forceinline__ void async_copy_b128(const unsigned short* g, unsigned short* l) {
  asm volatile("global_load_async_to_lds_b128 %0, %1, off"
               :: "v"((unsigned int)(unsigned long long)l), "v"(g)
               : "memory");
}
__device__ __forceinline__ void wait_async0() {
  asm volatile("s_wait_asynccnt 0x0" ::: "memory");
}

// A fragment: 16x32 bf16, lane = row M (half-waves split K 0-7/8-15, then +16)
__device__ __forceinline__ void load_a_frag(const unsigned short* sA, int msub, int lane,
                                            unsigned int* u) {
  const int m  = lane & 15;
  const int kb = (lane & 16) ? 8 : 0;
  const unsigned short* ar = sA + (msub * 16 + m) * STR;
#pragma unroll
  for (int j = 0; j < 4; ++j) {
    u[j]     = *(const unsigned int*)(ar + kb + 2 * j);
    u[4 + j] = *(const unsigned int*)(ar + 16 + kb + 2 * j);
  }
}

// B fragment: 32x16 bf16, lane = col N (half-waves split K 0-7/8-15, then +16)
__device__ __forceinline__ void load_b_frag(const unsigned short* sB, int col0, int lane,
                                            unsigned int* u) {
  const int kb = (lane & 16) ? 8 : 0;
  const unsigned short* br = sB + (col0 + (lane & 15)) * STR;
#pragma unroll
  for (int j = 0; j < 4; ++j) {
    u[j]     = *(const unsigned int*)(br + kb + 2 * j);
    u[4 + j] = *(const unsigned int*)(br + 16 + kb + 2 * j);
  }
}

// One K-step for a 32x32 per-wave tile: 2 A frags, 2 B frags, 4 WMMAs.
__device__ __forceinline__ void wmma_tile32(const unsigned short* sA, const unsigned short* sB,
                                            int lane, int wave, v8f c[2][2]) {
  union { v16bf v; unsigned int u[8]; } a0, a1, b;
  load_a_frag(sA, 0, lane, a0.u);
  load_a_frag(sA, 1, lane, a1.u);
#pragma unroll
  for (int ns = 0; ns < 2; ++ns) {
    load_b_frag(sB, wave * 32 + ns * 16, lane, b.u);
    c[0][ns] = __builtin_amdgcn_wmma_f32_16x16x32_bf16(false, a0.v, false, b.v, (short)0,
                                                       c[0][ns], false, false);
    c[1][ns] = __builtin_amdgcn_wmma_f32_16x16x32_bf16(false, a1.v, false, b.v, (short)0,
                                                       c[1][ns], false, false);
  }
}

__global__ void init_kernel(int* ints) {
  if (threadIdx.x < 18) ints[threadIdx.x] = 0;
}

__global__ __launch_bounds__(256) void router_kernel(
    const int* __restrict__ ids, const float* __restrict__ emb,
    const float* __restrict__ rw_w, const float* __restrict__ rb,
    unsigned short* __restrict__ xb, float* __restrict__ out_rw,
    int* __restrict__ out_se, int* __restrict__ ints) {
  const int t = blockIdx.x, tid = threadIdx.x;
  const int tok = ids[t];
  const float* er = emb + (size_t)tok * Hh;
  float acc[Ee];
#pragma unroll
  for (int e = 0; e < Ee; ++e) acc[e] = 0.f;
#pragma unroll
  for (int i = 0; i < 4; ++i) {
    const int h = tid * 4 + i;
    const float v = er[h];
    xb[(size_t)t * Hh + h] = f2bf(v);
#pragma unroll
    for (int e = 0; e < Ee; ++e) acc[e] += v * rw_w[h * Ee + e];
  }
  __shared__ float red[256][Ee];
#pragma unroll
  for (int e = 0; e < Ee; ++e) red[tid][e] = acc[e];
  __syncthreads();
  for (int s = 128; s > 0; s >>= 1) {
    if (tid < s)
#pragma unroll
      for (int e = 0; e < Ee; ++e) red[tid][e] += red[tid + s][e];
    __syncthreads();
  }
  if (tid == 0) {
    float lg[Ee];
#pragma unroll
    for (int e = 0; e < Ee; ++e) lg[e] = red[0][e] + rb[e];
    int i0 = 0;
    for (int e = 1; e < Ee; ++e) if (lg[e] > lg[i0]) i0 = e;      // first max
    int i1 = (i0 == 0) ? 1 : 0;
    for (int e = 0; e < Ee; ++e) if (e != i0 && lg[e] > lg[i1]) i1 = e;
    const float m = lg[i0];
    const float e0 = expf(lg[i0] - m), e1 = expf(lg[i1] - m);
    const float inv = 1.f / (e0 + e1);
    out_rw[t * 2 + 0] = e0 * inv; out_rw[t * 2 + 1] = e1 * inv;
    out_se[t * 2 + 0] = i0;       out_se[t * 2 + 1] = i1;
    const int p0 = atomicAdd(&ints[i0], 1);
    const int p1 = atomicAdd(&ints[i1], 1);
    ints[1024 + t * 2 + 0] = p0;  ints[1024 + t * 2 + 1] = p1;
  }
}

__global__ void prefix_kernel(int* ints) {
  if (threadIdx.x == 0 && blockIdx.x == 0) {
    int off = 0, nt = 0;
    for (int e = 0; e < Ee; ++e) {
      ints[8 + e] = off;
      const int c = ints[e];
      const int pad = (c + 31) & ~31;
      for (int r = 0; r < pad; r += 32) {
        ints[32 + nt]  = e;
        ints[296 + nt] = off + r;
        int vld = c - r; if (vld > 32) vld = 32;
        ints[560 + nt] = vld;
        ++nt;
      }
      off += pad;
    }
    ints[16] = off;
    ints[17] = nt;
  }
}

__global__ __launch_bounds__(256) void toklist_kernel(const int* __restrict__ out_se,
                                                      int* __restrict__ ints) {
  const int t = blockIdx.x * 256 + threadIdx.x;
#pragma unroll
  for (int k = 0; k < 2; ++k) {
    const int e = out_se[t * 2 + k];
    const int p = ints[1024 + t * 2 + k];
    ints[5120 + ints[8 + e] + p] = t;
  }
}

// h = gelu(Xg @ w1[e] + b1[e]) ; block tile M=32 (gathered tokens) x N=256, K=H
__global__ __launch_bounds__(256) void ffn1_kernel(
    const unsigned short* __restrict__ xb, const float* __restrict__ w1,
    const float* __restrict__ b1, unsigned short* __restrict__ hbuf,
    const int* __restrict__ ints) {
  const int tile = blockIdx.x;
  if (tile >= ints[17]) return;
  const int e = ints[32 + tile], row0 = ints[296 + tile], valid = ints[560 + tile];
  const int* tok_list = ints + 5120;
  const int n0 = blockIdx.y * 256;
  __shared__ __align__(16) unsigned short sA[32 * STR];
  __shared__ __align__(16) unsigned short sB[256 * STR];
  __shared__ int sTok[32];
  const int tid = threadIdx.x, wave = tid >> 5, lane = tid & 31;
  if (tid < 32) sTok[tid] = tok_list[row0 + tid];
  __syncthreads();
  const float* wmat = w1 + (size_t)e * Hh * Ii;
  v8f c[2][2] = {};
  const int ra = tid >> 2, ka = (tid & 3) << 3;   // A: row, 8-elem (16B) offset
  for (int k0 = 0; k0 < Hh; k0 += 32) {
    // A tile: 32x32 bf16, gathered token rows, async b128 (ASYNCcnt path)
    if (tid < 128 && ra < valid)
      async_copy_b128(xb + (size_t)sTok[ra] * Hh + k0 + ka, sA + ra * STR + ka);
#pragma unroll
    for (int i = 0; i < 8; ++i) { // B tile: 32x256 f32 -> bf16, transposed sB[n][k]
      const int q = tid + (i << 8);
      const int kk = q >> 6, n4 = (q & 63) << 2;
      const float4 w = *(const float4*)(wmat + (size_t)(k0 + kk) * Ii + n0 + n4);
      sB[(n4 + 0) * STR + kk] = f2bf(w.x);
      sB[(n4 + 1) * STR + kk] = f2bf(w.y);
      sB[(n4 + 2) * STR + kk] = f2bf(w.z);
      sB[(n4 + 3) * STR + kk] = f2bf(w.w);
    }
    if (k0 + 32 < Hh)
      __builtin_prefetch(wmat + (size_t)(k0 + 32 + (tid >> 6)) * Ii + n0 + ((tid & 63) << 2), 0, 1);
    wait_async0();
    __syncthreads();
    wmma_tile32(sA, sB, lane, wave, c);
    __syncthreads();
  }
#pragma unroll
  for (int ns = 0; ns < 2; ++ns) {
    const int gn = n0 + wave * 32 + ns * 16 + (lane & 15);
    const float bias = b1[e * Ii + gn];
#pragma unroll
    for (int ms = 0; ms < 2; ++ms)
#pragma unroll
      for (int v = 0; v < 8; ++v) {
        const int m = ms * 16 + v + ((lane & 16) ? 8 : 0);
        if (m < valid) {
          const float x = c[ms][ns][v] + bias;
          const float g = 0.5f * x * (1.0f + erff(x * 0.70710678f));
          hbuf[(size_t)(row0 + m) * Ii + gn] = f2bf(g);
        }
      }
  }
}

// contrib = h @ w2[e] + b2[e] ; K=I, per (token,expert) row, no atomics
__global__ __launch_bounds__(256) void ffn2_kernel(
    const unsigned short* __restrict__ hbuf, const float* __restrict__ w2,
    const float* __restrict__ b2, float* __restrict__ contrib,
    const int* __restrict__ ints) {
  const int tile = blockIdx.x;
  if (tile >= ints[17]) return;
  const int e = ints[32 + tile], row0 = ints[296 + tile], valid = ints[560 + tile];
  const int n0 = blockIdx.y * 256;
  __shared__ __align__(16) unsigned short sA[32 * STR];
  __shared__ __align__(16) unsigned short sB[256 * STR];
  const int tid = threadIdx.x, wave = tid >> 5, lane = tid & 31;
  const float* wmat = w2 + (size_t)e * Ii * Hh;
  v8f c[2][2] = {};
  const int ra = tid >> 2, ka = (tid & 3) << 3;
  for (int k0 = 0; k0 < Ii; k0 += 32) {
    if (tid < 128 && ra < valid)
      async_copy_b128(hbuf + (size_t)(row0 + ra) * Ii + k0 + ka, sA + ra * STR + ka);
#pragma unroll
    for (int i = 0; i < 8; ++i) {
      const int q = tid + (i << 8);
      const int kk = q >> 6, n4 = (q & 63) << 2;
      const float4 w = *(const float4*)(wmat + (size_t)(k0 + kk) * Hh + n0 + n4);
      sB[(n4 + 0) * STR + kk] = f2bf(w.x);
      sB[(n4 + 1) * STR + kk] = f2bf(w.y);
      sB[(n4 + 2) * STR + kk] = f2bf(w.z);
      sB[(n4 + 3) * STR + kk] = f2bf(w.w);
    }
    if (k0 + 32 < Ii)
      __builtin_prefetch(wmat + (size_t)(k0 + 32 + (tid >> 6)) * Hh + n0 + ((tid & 63) << 2), 0, 1);
    wait_async0();
    __syncthreads();
    wmma_tile32(sA, sB, lane, wave, c);
    __syncthreads();
  }
#pragma unroll
  for (int ns = 0; ns < 2; ++ns) {
    const int gn = n0 + wave * 32 + ns * 16 + (lane & 15);
    const float bias = b2[e * Hh + gn];
#pragma unroll
    for (int ms = 0; ms < 2; ++ms)
#pragma unroll
      for (int v = 0; v < 8; ++v) {
        const int m = ms * 16 + v + ((lane & 16) ? 8 : 0);
        if (m < valid) contrib[(size_t)(row0 + m) * Hh + gn] = c[ms][ns][v] + bias;
      }
  }
}

// hidden[t] = contrib[row(t,0)] + contrib[row(t,1)]  (exact two experts per token)
__global__ __launch_bounds__(256) void combine_kernel(
    const float* __restrict__ contrib, const int* __restrict__ out_se,
    const int* __restrict__ ints, unsigned short* __restrict__ eobf) {
  const int idx = blockIdx.x * 256 + threadIdx.x;
  const int t = idx >> 10, h = idx & (Hh - 1);
  const int e0 = out_se[t * 2], e1 = out_se[t * 2 + 1];
  const int r0 = ints[8 + e0] + ints[1024 + t * 2];
  const int r1 = ints[8 + e1] + ints[1024 + t * 2 + 1];
  const float v = contrib[(size_t)r0 * Hh + h] + contrib[(size_t)r1 * Hh + h];
  eobf[idx] = f2bf(v);
}

// logits = hidden @ lm_w + lm_b ; block tile M=32 x N=256, K=H
__global__ __launch_bounds__(256) void lm_kernel(
    const unsigned short* __restrict__ eobf, const float* __restrict__ lm_w,
    const float* __restrict__ lm_b, float* __restrict__ out) {
  const int t0 = blockIdx.x * 32;
  const int n0 = blockIdx.y * 256;
  __shared__ __align__(16) unsigned short sA[32 * STR];
  __shared__ __align__(16) unsigned short sB[256 * STR];
  const int tid = threadIdx.x, wave = tid >> 5, lane = tid & 31;
  v8f c[2][2] = {};
  const int ra = tid >> 2, ka = (tid & 3) << 3;
  for (int k0 = 0; k0 < Hh; k0 += 32) {
    if (tid < 128)
      async_copy_b128(eobf + (size_t)(t0 + ra) * Hh + k0 + ka, sA + ra * STR + ka);
#pragma unroll
    for (int i = 0; i < 8; ++i) {
      const int q = tid + (i << 8);
      const int kk = q >> 6, n4 = (q & 63) << 2;
      const float4 w = *(const float4*)(lm_w + (size_t)(k0 + kk) * Vv + n0 + n4);
      sB[(n4 + 0) * STR + kk] = f2bf(w.x);
      sB[(n4 + 1) * STR + kk] = f2bf(w.y);
      sB[(n4 + 2) * STR + kk] = f2bf(w.z);
      sB[(n4 + 3) * STR + kk] = f2bf(w.w);
    }
    if (k0 + 32 < Hh)
      __builtin_prefetch(lm_w + (size_t)(k0 + 32 + (tid >> 6)) * Vv + n0 + ((tid & 63) << 2), 0, 1);
    wait_async0();
    __syncthreads();
    wmma_tile32(sA, sB, lane, wave, c);
    __syncthreads();
  }
#pragma unroll
  for (int ns = 0; ns < 2; ++ns) {
    const int gn = n0 + wave * 32 + ns * 16 + (lane & 15);
    const float bias = lm_b[gn];
#pragma unroll
    for (int ms = 0; ms < 2; ++ms)
#pragma unroll
      for (int v = 0; v < 8; ++v) {
        const int m = ms * 16 + v + ((lane & 16) ? 8 : 0);
        out[(size_t)(t0 + m) * Vv + gn] = c[ms][ns][v] + bias;
      }
  }
}

extern "C" void kernel_launch(void* const* d_in, const int* in_sizes, int n_in,
                              void* d_out, int out_size, void* d_ws, size_t ws_size,
                              hipStream_t stream) {
  const int*   ids      = (const int*)d_in[0];
  const float* emb      = (const float*)d_in[1];
  const float* router_w = (const float*)d_in[2];
  const float* router_b = (const float*)d_in[3];
  const float* w1       = (const float*)d_in[4];
  const float* b1       = (const float*)d_in[5];
  const float* w2       = (const float*)d_in[6];
  const float* b2       = (const float*)d_in[7];
  const float* lm_w     = (const float*)d_in[8];
  const float* lm_b     = (const float*)d_in[9];

  float* logits = (float*)d_out;
  float* out_rw = logits + (size_t)Tn * Vv;
  int*   out_se = (int*)d_out + (size_t)Tn * Vv + (size_t)Tn * 2;

  char* ws = (char*)d_ws;
  unsigned short* xb      = (unsigned short*)(ws);              // [T,H] bf16      4.19 MB
  unsigned short* hbuf    = (unsigned short*)(ws + 4194304);    // [4352,I] bf16  35.65 MB
  float*          contrib = (float*)(ws + 39845888);            // [4352,H] f32   17.83 MB
  unsigned short* eobf    = (unsigned short*)(ws + 57671680);   // [T,H] bf16      4.19 MB
  int*            ints    = (int*)(ws + 61865984);              // ~38 KB

  init_kernel<<<1, 32, 0, stream>>>(ints);
  router_kernel<<<Tn, 256, 0, stream>>>(ids, emb, router_w, router_b, xb, out_rw, out_se, ints);
  prefix_kernel<<<1, 1, 0, stream>>>(ints);
  toklist_kernel<<<Tn / 256, 256, 0, stream>>>(out_se, ints);
  ffn1_kernel<<<dim3(MAX_T32, Ii / 256), 256, 0, stream>>>(xb, w1, b1, hbuf, ints);
  ffn2_kernel<<<dim3(MAX_T32, Hh / 256), 256, 0, stream>>>(hbuf, w2, b2, contrib, ints);
  combine_kernel<<<(Tn * Hh) / 256, 256, 0, stream>>>(contrib, out_se, ints, eobf);
  lm_kernel<<<dim3(Tn / 32, Vv / 256), 256, 0, stream>>>(eobf, lm_w, lm_b, logits);
}
